// Encoder_72224170050035
// MI455X (gfx1250) — matrix operand order repaired and checked by reading, not compiled, not run on hardware
//
#include <hip/hip_runtime.h>
#include <hip/hip_bf16.h>

typedef __bf16 bf16_t;
typedef __attribute__((ext_vector_type(16))) __bf16 v16bf;
typedef __attribute__((ext_vector_type(8)))  __bf16 v8bf;
typedef __attribute__((ext_vector_type(8)))  float  v8f;

#define AS_STRIDE 40   // bf16 elements; 80B row stride (16B aligned)
#define BS_STRIDE 40

// ---------------------------------------------------------------------------
// Large-tile bf16 WMMA GEMM: C[M,N] = A[M,K] * BT[N,K]^T (+ bias[n])
// 256 threads / 8 waves, 128x128 block tile, 32x64 register tile per wave.
// Software-pipelined: next K-tile global loads issued ahead of the WMMA block.
// Batched over blockIdx.z with (zb = z/zdiv, zh = z%zdiv) offsets per operand.
// ---------------------------------------------------------------------------
__launch_bounds__(256)
__global__ void gemm_bf16_wmma_128(
    const bf16_t* __restrict__ A, const bf16_t* __restrict__ BT,
    float* __restrict__ Cf, bf16_t* __restrict__ Cb,
    const float* __restrict__ bias,
    int M, int N, int K, int lda, int ldbt, int ldc,
    int zdiv,
    long long sAb, long long sAh,
    long long sBb, long long sBh,
    long long sCb, long long sCh)
{
  __shared__ __attribute__((aligned(16))) bf16_t AS[128 * AS_STRIDE];
  __shared__ __attribute__((aligned(16))) bf16_t BS[128 * BS_STRIDE];

  const int tid  = threadIdx.x;
  const int wave = tid >> 5;
  const int lane = tid & 31;

  const int z  = blockIdx.z;
  const int zb = z / zdiv;
  const int zh = z % zdiv;
  const bf16_t* Ab = A  + (long long)zb * sAb + (long long)zh * sAh;
  const bf16_t* Bb = BT + (long long)zb * sBb + (long long)zh * sBh;
  const long long offC = (long long)zb * sCb + (long long)zh * sCh;

  const int m0 = blockIdx.y * 128;
  const int n0 = blockIdx.x * 128;

  // staging coordinates: 2 x v8bf per operand per thread
  const int sr0 = (tid * 8) >> 5;          // rows 0..63   (p=0)
  const int sc  = (tid * 8) & 31;
  const int sr1 = sr0 + 64;                // rows 64..127 (p=1)

  const int wm = (wave & 3) * 32;          // wave row base within tile
  const int wn = (wave >> 2) * 64;         // wave col base within tile

  const int r  = lane & 15;
  const int hi = lane >> 4;

  v8f acc[2][4] = {};

  // ---- preload K-tile 0 into registers, then LDS ----
  v8bf a0 = *(const v8bf*)(Ab + (long long)(m0 + sr0) * lda  + sc);
  v8bf a1 = *(const v8bf*)(Ab + (long long)(m0 + sr1) * lda  + sc);
  v8bf b0 = *(const v8bf*)(Bb + (long long)(n0 + sr0) * ldbt + sc);
  v8bf b1 = *(const v8bf*)(Bb + (long long)(n0 + sr1) * ldbt + sc);
  *(v8bf*)(AS + sr0 * AS_STRIDE + sc) = a0;
  *(v8bf*)(AS + sr1 * AS_STRIDE + sc) = a1;
  *(v8bf*)(BS + sr0 * BS_STRIDE + sc) = b0;
  *(v8bf*)(BS + sr1 * BS_STRIDE + sc) = b1;
  __syncthreads();

  for (int kk = 0; kk < K; kk += 32) {
    const bool more = (kk + 32) < K;
    // ---- issue next tile's global loads early (hides HBM latency) ----
    if (more) {
      int k2 = kk + 32;
      a0 = *(const v8bf*)(Ab + (long long)(m0 + sr0) * lda  + k2 + sc);
      a1 = *(const v8bf*)(Ab + (long long)(m0 + sr1) * lda  + k2 + sc);
      b0 = *(const v8bf*)(Bb + (long long)(n0 + sr0) * ldbt + k2 + sc);
      b1 = *(const v8bf*)(Bb + (long long)(n0 + sr1) * ldbt + k2 + sc);
    }

    // ---- A fragments: rows wm + mi*16 + r ; K halves by lane hi bit ----
    v16bf af[2];
#pragma unroll
    for (int mi = 0; mi < 2; ++mi) {
      const bf16_t* ap = AS + (wm + mi * 16 + r) * AS_STRIDE + hi * 8;
      v8bf alo = *(const v8bf*)(ap);
      v8bf ahi = *(const v8bf*)(ap + 16);
#pragma unroll
      for (int i = 0; i < 8; ++i) { af[mi][i] = alo[i]; af[mi][8 + i] = ahi[i]; }
    }

#pragma unroll
    for (int nb = 0; nb < 4; ++nb) {
      const bf16_t* bp = BS + (wn + nb * 16 + r) * BS_STRIDE + hi * 8;
      v8bf blo = *(const v8bf*)(bp);
      v8bf bhi = *(const v8bf*)(bp + 16);
      v16bf bfrag;
#pragma unroll
      for (int i = 0; i < 8; ++i) { bfrag[i] = blo[i]; bfrag[8 + i] = bhi[i]; }
#pragma unroll
      for (int mi = 0; mi < 2; ++mi) {
        acc[mi][nb] = __builtin_amdgcn_wmma_f32_16x16x32_bf16(
            false, af[mi], false, bfrag, (short)0, acc[mi][nb], false, false);
      }
    }

    if (more) {
      __syncthreads();   // everyone done reading current LDS tiles
      *(v8bf*)(AS + sr0 * AS_STRIDE + sc) = a0;
      *(v8bf*)(AS + sr1 * AS_STRIDE + sc) = a1;
      *(v8bf*)(BS + sr0 * BS_STRIDE + sc) = b0;
      *(v8bf*)(BS + sr1 * BS_STRIDE + sc) = b1;
      __syncthreads();
    }
  }

  // ---- epilogue: C layout — element e: lane<16 -> M=e, lane>=16 -> M=8+e ----
  const int c     = lane & 15;
  const int rbase = hi * 8;
#pragma unroll
  for (int nb = 0; nb < 4; ++nb) {
    int gn   = n0 + wn + nb * 16 + c;
    float bv = bias ? bias[gn] : 0.0f;
#pragma unroll
    for (int mi = 0; mi < 2; ++mi) {
#pragma unroll
      for (int e = 0; e < 8; ++e) {
        int gm = m0 + wm + mi * 16 + rbase + e;
        long long cidx = offC + (long long)gm * ldc + gn;
        float v = acc[mi][nb][e] + bv;
        if (Cf) Cf[cidx] = v;
        if (Cb) Cb[cidx] = (bf16_t)v;
      }
    }
  }
}

// ---------------------------------------------------------------------------
// Small-tile (64x64) bf16 WMMA GEMM — used for att @ V (N = 64).
// 128 threads / 4 waves, 16x64 register tile per wave.
// ---------------------------------------------------------------------------
__launch_bounds__(128)
__global__ void gemm_bf16_wmma_64(
    const bf16_t* __restrict__ A, const bf16_t* __restrict__ BT,
    float* __restrict__ Cf, bf16_t* __restrict__ Cb,
    const float* __restrict__ bias,
    int M, int N, int K, int lda, int ldbt, int ldc,
    int zdiv,
    long long sAb, long long sAh,
    long long sBb, long long sBh,
    long long sCb, long long sCh)
{
  __shared__ __attribute__((aligned(16))) bf16_t AS[64 * AS_STRIDE];
  __shared__ __attribute__((aligned(16))) bf16_t BS[64 * BS_STRIDE];

  const int tid  = threadIdx.x;
  const int wave = tid >> 5;
  const int lane = tid & 31;

  const int z  = blockIdx.z;
  const int zb = z / zdiv;
  const int zh = z % zdiv;
  const bf16_t* Ab = A  + (long long)zb * sAb + (long long)zh * sAh;
  const bf16_t* Bb = BT + (long long)zb * sBb + (long long)zh * sBh;
  const long long offC = (long long)zb * sCb + (long long)zh * sCh;

  const int m0 = blockIdx.y * 64;
  const int n0 = blockIdx.x * 64;

  v8f acc[4] = {};
  const int r  = lane & 15;
  const int hi = lane >> 4;

  for (int kk = 0; kk < K; kk += 32) {
#pragma unroll
    for (int p = 0; p < 2; ++p) {
      int idx8 = (p * 128 + tid) * 8;
      int row  = idx8 >> 5;
      int col  = idx8 & 31;
      v8bf va = *(const v8bf*)(Ab + (long long)(m0 + row) * lda + kk + col);
      *(v8bf*)(AS + row * AS_STRIDE + col) = va;
      v8bf vb = *(const v8bf*)(Bb + (long long)(n0 + row) * ldbt + kk + col);
      *(v8bf*)(BS + row * BS_STRIDE + col) = vb;
    }
    __syncthreads();

    const bf16_t* ap = AS + (wave * 16 + r) * AS_STRIDE + hi * 8;
    v8bf alo = *(const v8bf*)(ap);
    v8bf ahi = *(const v8bf*)(ap + 16);
    v16bf afrag;
#pragma unroll
    for (int i = 0; i < 8; ++i) { afrag[i] = alo[i]; afrag[8 + i] = ahi[i]; }

#pragma unroll
    for (int nb = 0; nb < 4; ++nb) {
      const bf16_t* bp = BS + (nb * 16 + r) * BS_STRIDE + hi * 8;
      v8bf blo = *(const v8bf*)(bp);
      v8bf bhi = *(const v8bf*)(bp + 16);
      v16bf bfrag;
#pragma unroll
      for (int i = 0; i < 8; ++i) { bfrag[i] = blo[i]; bfrag[8 + i] = bhi[i]; }
      acc[nb] = __builtin_amdgcn_wmma_f32_16x16x32_bf16(
          false, afrag, false, bfrag, (short)0, acc[nb], false, false);
    }
    __syncthreads();
  }

  const int c     = lane & 15;
  const int rbase = hi * 8;
#pragma unroll
  for (int nb = 0; nb < 4; ++nb) {
    int gn   = n0 + nb * 16 + c;
    float bv = bias ? bias[gn] : 0.0f;
#pragma unroll
    for (int e = 0; e < 8; ++e) {
      int gm = m0 + wave * 16 + rbase + e;
      long long cidx = offC + (long long)gm * ldc + gn;
      float v = acc[nb][e] + bv;
      if (Cf) Cf[cidx] = v;
      if (Cb) Cb[cidx] = (bf16_t)v;
    }
  }
}

// ---------------------------------------------------------------------------
// Weight packing kernels (fp32 -> bf16, transposed to [N,K])
// ---------------------------------------------------------------------------
__global__ void pack_qkv(const float* __restrict__ Wq, const float* __restrict__ Wk,
                         const float* __restrict__ Wv, bf16_t* __restrict__ out)
{
  // out[(l*1536 + n)*512 + d],  n = sel*512 + h*64 + dk
  long long i = (long long)blockIdx.x * 256 + threadIdx.x;
  if (i >= 6LL * 1536 * 512) return;
  int d = (int)(i & 511);
  long long t = i >> 9;
  int n = (int)(t % 1536);
  int l = (int)(t / 1536);
  int sel = n >> 9;
  int nn  = n & 511;
  int h   = nn >> 6;
  int dk  = nn & 63;
  const float* W = (sel == 0) ? Wq : (sel == 1) ? Wk : Wv;
  float v = W[(((long long)l * 8 + h) * 512 + d) * 64 + dk];
  out[i] = (bf16_t)v;
}

__global__ void pack_T(const float* __restrict__ W, bf16_t* __restrict__ out,
                       int K, int N)
{
  // out[(l*N + n)*K + k] = W[(l*K + k)*N + n]   (L = 6)
  long long i = (long long)blockIdx.x * 256 + threadIdx.x;
  long long total = 6LL * K * N;
  if (i >= total) return;
  int k = (int)(i % K);
  long long t = i / K;
  int n = (int)(t % N);
  int l = (int)(t / N);
  out[i] = (bf16_t)W[((long long)l * K + k) * N + n];
}

// V (inside qkv buffer) -> Vt[bh][dk][s]
__global__ void transpose_v(const bf16_t* __restrict__ qkv, bf16_t* __restrict__ Vt)
{
  long long i = (long long)blockIdx.x * 256 + threadIdx.x;
  if (i >= 32LL * 8 * 512 * 64) return;
  int dk = (int)(i & 63);
  long long t = i >> 6;
  int s  = (int)(t & 511);
  int bh = (int)(t >> 9);
  int b  = bh >> 3, h = bh & 7;
  bf16_t v = qkv[((long long)(b * 512 + s)) * 1536 + 1024 + h * 64 + dk];
  Vt[((long long)bh * 64 + dk) * 512 + s] = v;
}

// ---------------------------------------------------------------------------
// Embedding + sinusoidal positional encoding
// ---------------------------------------------------------------------------
__global__ void embed_kernel(const int* __restrict__ tokens, const float* __restrict__ emb,
                             float* __restrict__ xf, bf16_t* __restrict__ xb)
{
  int row = blockIdx.x;          // b*S + s
  int s   = row & 511;
  int tok = tokens[row];
  const float* e = emb + (long long)tok * 512;
#pragma unroll
  for (int i = 0; i < 4; ++i) {
    int d = threadIdx.x + i * 128;
    int pair = d >> 1;
    float div = __expf(-logf(10000.0f) * (float)(2 * pair) / 512.0f);
    float ang = (float)s * div;
    float pe  = (d & 1) ? __cosf(ang) : __sinf(ang);
    float v = e[d] + pe;
    xf[(long long)row * 512 + d] = v;
    xb[(long long)row * 512 + d] = (bf16_t)v;
  }
}

// ---------------------------------------------------------------------------
// In-place softmax over 512-wide rows of bf16 scores (fp32 math)
// ---------------------------------------------------------------------------
__global__ void softmax_kernel(bf16_t* __restrict__ scores, float scale)
{
  __shared__ float red[128];
  long long row = blockIdx.x;
  bf16_t* p = scores + row * 512;
  int tid = threadIdx.x;

  float v[4];
  float m = -1e30f;
#pragma unroll
  for (int i = 0; i < 4; ++i) {
    v[i] = (float)p[tid + i * 128] * scale;
    m = fmaxf(m, v[i]);
  }
  red[tid] = m; __syncthreads();
  for (int o = 64; o > 0; o >>= 1) {
    if (tid < o) red[tid] = fmaxf(red[tid], red[tid + o]);
    __syncthreads();
  }
  m = red[0]; __syncthreads();

  float sum = 0.0f;
#pragma unroll
  for (int i = 0; i < 4; ++i) { v[i] = __expf(v[i] - m); sum += v[i]; }
  red[tid] = sum; __syncthreads();
  for (int o = 64; o > 0; o >>= 1) {
    if (tid < o) red[tid] += red[tid + o];
    __syncthreads();
  }
  sum = red[0]; __syncthreads();

  float inv = 1.0f / sum;
#pragma unroll
  for (int i = 0; i < 4; ++i) p[tid + i * 128] = (bf16_t)(v[i] * inv);
}

// ---------------------------------------------------------------------------
// LayerNorm with fused residual add: y = LN(inp + res)*g + b
// writes fp32 (new x) and bf16 (gemm input) copies
// ---------------------------------------------------------------------------
__global__ void ln_kernel(const float* __restrict__ inp, const float* __restrict__ res,
                          const float* __restrict__ g, const float* __restrict__ b,
                          float* __restrict__ outf, bf16_t* __restrict__ outb)
{
  __shared__ float red[128];
  long long row = blockIdx.x;
  const float* ip = inp + row * 512;
  const float* rp = res + row * 512;
  int tid = threadIdx.x;

  float t[4];
  float s = 0.0f;
#pragma unroll
  for (int i = 0; i < 4; ++i) {
    int d = tid + i * 128;
    t[i] = ip[d] + rp[d];
    s += t[i];
  }
  red[tid] = s; __syncthreads();
  for (int o = 64; o > 0; o >>= 1) {
    if (tid < o) red[tid] += red[tid + o];
    __syncthreads();
  }
  float mu = red[0] * (1.0f / 512.0f); __syncthreads();

  float vs = 0.0f;
#pragma unroll
  for (int i = 0; i < 4; ++i) { float d0 = t[i] - mu; vs += d0 * d0; }
  red[tid] = vs; __syncthreads();
  for (int o = 64; o > 0; o >>= 1) {
    if (tid < o) red[tid] += red[tid + o];
    __syncthreads();
  }
  float var = red[0] * (1.0f / 512.0f);
  float inv = rsqrtf(var + 1e-5f);

#pragma unroll
  for (int i = 0; i < 4; ++i) {
    int d = tid + i * 128;
    float y = (t[i] - mu) * inv * g[d] + b[d];
    outf[row * 512 + d] = y;
    outb[row * 512 + d] = (bf16_t)y;
  }
}

// ---------------------------------------------------------------------------
// Host launcher
// ---------------------------------------------------------------------------
extern "C" void kernel_launch(void* const* d_in, const int* in_sizes, int n_in,
                              void* d_out, int out_size, void* d_ws, size_t ws_size,
                              hipStream_t stream)
{
  (void)in_sizes; (void)n_in; (void)out_size; (void)ws_size;

  const int*   tokens = (const int*)  d_in[0];
  const float* emb    = (const float*)d_in[1];
  const float* Wq     = (const float*)d_in[2];
  const float* Wk     = (const float*)d_in[3];
  const float* Wv     = (const float*)d_in[4];
  const float* Wo     = (const float*)d_in[5];
  const float* bo     = (const float*)d_in[6];
  const float* ln1_g  = (const float*)d_in[7];
  const float* ln1_b  = (const float*)d_in[8];
  const float* ln2_g  = (const float*)d_in[9];
  const float* ln2_b  = (const float*)d_in[10];
  const float* W1     = (const float*)d_in[11];
  const float* b1     = (const float*)d_in[12];
  const float* W2     = (const float*)d_in[13];
  const float* b2     = (const float*)d_in[14];

  char* ws = (char*)d_ws;
  size_t off = 0;
  auto alloc = [&](size_t bytes) -> void* {
    void* p = ws + off;
    off += (bytes + 255) & ~(size_t)255;
    return p;
  };

  bf16_t* WqkvT  = (bf16_t*)alloc(6LL * 1536 * 512 * 2);
  bf16_t* WoT    = (bf16_t*)alloc(6LL * 512 * 512 * 2);
  bf16_t* W1T    = (bf16_t*)alloc(6LL * 2048 * 512 * 2);
  bf16_t* W2T    = (bf16_t*)alloc(6LL * 512 * 2048 * 2);
  float*  xf     = (float*) alloc(16384LL * 512 * 4);
  bf16_t* xb     = (bf16_t*)alloc(16384LL * 512 * 2);
  bf16_t* qkv    = (bf16_t*)alloc(16384LL * 1536 * 2);
  bf16_t* Vt     = (bf16_t*)alloc(256LL * 64 * 512 * 2);
  bf16_t* scores = (bf16_t*)alloc(256LL * 512 * 512 * 2);
  bf16_t* h1     = scores;  // alias: scores dead by the time FFN runs
  bf16_t* attout = (bf16_t*)alloc(16384LL * 512 * 2);
  float*  tmp    = (float*) alloc(16384LL * 512 * 4);

  // ---- pack weights to bf16, transposed [N,K] ----
  pack_qkv<<<(unsigned)((6LL * 1536 * 512 + 255) / 256), 256, 0, stream>>>(Wq, Wk, Wv, WqkvT);
  pack_T  <<<(unsigned)((6LL * 512 * 512  + 255) / 256), 256, 0, stream>>>(Wo, WoT, 512, 512);
  pack_T  <<<(unsigned)((6LL * 2048 * 512 + 255) / 256), 256, 0, stream>>>(W1, W1T, 512, 2048);
  pack_T  <<<(unsigned)((6LL * 512 * 2048 + 255) / 256), 256, 0, stream>>>(W2, W2T, 2048, 512);

  // ---- embedding + positional encoding ----
  embed_kernel<<<16384, 128, 0, stream>>>(tokens, emb, xf, xb);

  for (int l = 0; l < 6; ++l) {
    // QKV projection: [16384,512] x [512,1536]
    gemm_bf16_wmma_128<<<dim3(1536 / 128, 16384 / 128, 1), 256, 0, stream>>>(
        xb, WqkvT + (long long)l * 1536 * 512, nullptr, qkv, nullptr,
        16384, 1536, 512, 512, 512, 1536,
        1, 0, 0, 0, 0, 0, 0);

    transpose_v<<<(unsigned)((32LL * 8 * 512 * 64 + 255) / 256), 256, 0, stream>>>(qkv, Vt);

    // energy = Q K^T per (b,h): [512,512] = [512,64] x [64,512]
    gemm_bf16_wmma_128<<<dim3(512 / 128, 512 / 128, 256), 256, 0, stream>>>(
        qkv, qkv + 512, nullptr, scores, nullptr,
        512, 512, 64, 1536, 1536, 512,
        8, 512LL * 1536, 64, 512LL * 1536, 64, 8LL * 512 * 512, 512LL * 512);

    softmax_kernel<<<256 * 512, 128, 0, stream>>>(scores, 0.125f);

    // head = att V per (b,h): [512,64] = [512,512] x [512,64]  (N=64 -> 64-tile)
    gemm_bf16_wmma_64<<<dim3(1, 512 / 64, 256), 128, 0, stream>>>(
        scores, Vt, nullptr, attout, nullptr,
        512, 64, 512, 512, 512, 512,
        8, 8LL * 512 * 512, 512LL * 512, 8LL * 64 * 512, 64LL * 512, 512LL * 512, 64);

    // output projection: [16384,512] x [512,512] + bo
    gemm_bf16_wmma_128<<<dim3(512 / 128, 16384 / 128, 1), 256, 0, stream>>>(
        attout, WoT + (long long)l * 512 * 512, tmp, nullptr, bo + l * 512,
        16384, 512, 512, 512, 512, 512,
        1, 0, 0, 0, 0, 0, 0);

    ln_kernel<<<16384, 128, 0, stream>>>(tmp, xf, ln1_g + l * 512, ln1_b + l * 512, xf, xb);

    // FFN1: [16384,2048] = x W1 + b1 (no activation in reference)
    gemm_bf16_wmma_128<<<dim3(2048 / 128, 16384 / 128, 1), 256, 0, stream>>>(
        xb, W1T + (long long)l * 2048 * 512, nullptr, h1, b1 + l * 2048,
        16384, 2048, 512, 512, 512, 2048,
        1, 0, 0, 0, 0, 0, 0);

    // FFN2: [16384,512] = h1 W2 + b2
    gemm_bf16_wmma_128<<<dim3(512 / 128, 16384 / 128, 1), 256, 0, stream>>>(
        h1, W2T + (long long)l * 512 * 2048, tmp, nullptr, b2 + l * 512,
        16384, 512, 2048, 2048, 2048, 512,
        1, 0, 0, 0, 0, 0, 0);

    float* outf = (l == 5) ? (float*)d_out : xf;
    ln_kernel<<<16384, 128, 0, stream>>>(tmp, xf, ln2_g + l * 512, ln2_b + l * 512, outf, xb);
  }
}
